// RecurrentPPOAgent_64080912056814
// MI455X (gfx1250) — matrix operand order repaired and checked by reading, not compiled, not run on hardware
//
#include <hip/hip_runtime.h>
#include <hip/hip_bf16.h>

typedef __attribute__((ext_vector_type(16))) _Float16 v16h;
typedef __attribute__((ext_vector_type(8)))  _Float16 v8h;
typedef __attribute__((ext_vector_type(8)))  float    v8f;

#define T_DIM 512
#define B_DIM 1024
#define D_DIM 128
#define H_DIM 64
#define A_DIM 18
#define G_DIM 192   // 3*H

// ---------------- WMMA helpers (gfx1250 wave32) ----------------
// D = A(16x32 f16) x B(32x16 f16) + C(16x16 f32)
__device__ __forceinline__ v8f wmma16(v16h a, v16h b, v8f c) {
  return __builtin_amdgcn_wmma_f32_16x16x32_f16(
      /*neg_a=*/false, a, /*neg_b=*/false, b,
      /*c_mod=*/(short)0, c, /*reuse_a=*/false, /*reuse_b=*/false);
}

// A fragment (16x32, MxK) from global f32 row-major:
// lanes 0-15: row=lane, K = k0+0..7 (elems 0-7) and k0+16..23 (elems 8-15)
// lanes 16-31: row=lane-16, K = k0+8..15 and k0+24..31
__device__ __forceinline__ v16h load_a_global_f32(const float* base, int ld, int k0) {
  const int lane = threadIdx.x & 31;
  const int m    = lane & 15;
  const int ko   = k0 + ((lane >> 4) << 3);
  const float* p = base + m * ld + ko;
  v16h a;
#pragma unroll
  for (int i = 0; i < 8; ++i) a[i]     = (_Float16)p[i];
#pragma unroll
  for (int i = 0; i < 8; ++i) a[8 + i] = (_Float16)p[16 + i];
  return a;
}

__device__ __forceinline__ v16h load_a_lds(const _Float16* base, int ld, int k0) {
  const int lane = threadIdx.x & 31;
  const int m    = lane & 15;
  const int ko   = k0 + ((lane >> 4) << 3);
  const _Float16* p = base + m * ld + ko;
  v16h a;
#pragma unroll
  for (int i = 0; i < 8; ++i) a[i]     = p[i];
#pragma unroll
  for (int i = 0; i < 8; ++i) a[8 + i] = p[16 + i];
  return a;
}

// B fragment (32x16, KxN) for out = x @ W.T with W stored row-major [N][K]:
// B[k][n] = W[n][k] -> lane n reads K = k0+8h+0..7 and k0+16+8h+0..7 of W row n.
__device__ __forceinline__ v16h load_b_lds(const _Float16* W, int ldk, int n0, int k0) {
  const int lane = threadIdx.x & 31;
  const _Float16* p = W + (n0 + (lane & 15)) * ldk + k0 + ((lane >> 4) << 3);
  v16h b;
#pragma unroll
  for (int i = 0; i < 8; ++i) b[i] = p[i];
#pragma unroll
  for (int i = 0; i < 8; ++i) b[8 + i] = p[16 + i];
  return b;
}

// =====================================================================
// Kernel 1: fused MLP (2x Linear+ReLU) + gi = x @ Wih.T + bih over all
// T*B rows, both branches. Each wave owns a 16-row pipeline.
// =====================================================================
__global__ __launch_bounds__(128) void rppo_mlp_gi_kernel(
    const float* __restrict__ obs,
    const float* __restrict__ aW1, const float* __restrict__ ab1,
    const float* __restrict__ aW2, const float* __restrict__ ab2,
    const float* __restrict__ aWih, const float* __restrict__ abih,
    const float* __restrict__ cW1, const float* __restrict__ cb1,
    const float* __restrict__ cW2, const float* __restrict__ cb2,
    const float* __restrict__ cWih, const float* __restrict__ cbih,
    _Float16* __restrict__ giA, _Float16* __restrict__ giC)
{
  __shared__ _Float16 sW1[H_DIM * D_DIM];    // 16 KB
  __shared__ _Float16 sW2[H_DIM * H_DIM];    //  8 KB
  __shared__ _Float16 sWih[G_DIM * H_DIM];   // 24 KB
  __shared__ float    sb1[H_DIM], sb2[H_DIM], sbih[G_DIM];
  __shared__ __align__(16) _Float16 sX[4][16 * H_DIM];   // per-wave scratch, 8 KB

  const int tid    = threadIdx.x;
  const int branch = blockIdx.y;
  const float* W1  = branch ? cW1  : aW1;
  const float* b1  = branch ? cb1  : ab1;
  const float* W2  = branch ? cW2  : aW2;
  const float* b2  = branch ? cb2  : ab2;
  const float* Wih = branch ? cWih : aWih;
  const float* bih = branch ? cbih : abih;
  _Float16*    gi  = branch ? giC  : giA;

  for (int i = tid; i < H_DIM * D_DIM; i += 128) sW1[i]  = (_Float16)W1[i];
  for (int i = tid; i < H_DIM * H_DIM; i += 128) sW2[i]  = (_Float16)W2[i];
  for (int i = tid; i < G_DIM * H_DIM; i += 128) sWih[i] = (_Float16)Wih[i];
  if (tid < H_DIM) { sb1[tid] = b1[tid]; sb2[tid] = b2[tid]; }
  for (int i = tid; i < G_DIM; i += 128) sbih[i] = bih[i];
  __syncthreads();

  const int wave = tid >> 5;
  const int lane = tid & 31;
  const int crow = (lane >> 4) << 3;   // C/D frag: row base 0 or 8
  const int ccol = lane & 15;          // C/D frag: column
  const long row0 = ((long)blockIdx.x * 4 + wave) * 16;
  const float* obsp = obs + row0 * D_DIM;
  _Float16* xb = sX[wave];

  // ---- layer 1: x1 = relu(obs @ W1.T + b1), K=128 ----
  v16h a0 = load_a_global_f32(obsp, D_DIM, 0);
  v16h a1 = load_a_global_f32(obsp, D_DIM, 32);
  v16h a2 = load_a_global_f32(obsp, D_DIM, 64);
  v16h a3 = load_a_global_f32(obsp, D_DIM, 96);
#pragma unroll
  for (int nt = 0; nt < 4; ++nt) {
    v8f acc = {};
    acc = wmma16(a0, load_b_lds(sW1, D_DIM, nt * 16, 0),  acc);
    acc = wmma16(a1, load_b_lds(sW1, D_DIM, nt * 16, 32), acc);
    acc = wmma16(a2, load_b_lds(sW1, D_DIM, nt * 16, 64), acc);
    acc = wmma16(a3, load_b_lds(sW1, D_DIM, nt * 16, 96), acc);
    const float bias = sb1[nt * 16 + ccol];
#pragma unroll
    for (int i = 0; i < 8; ++i) {
      float v = acc[i] + bias;
      v = v > 0.0f ? v : 0.0f;
      xb[(crow + i) * H_DIM + nt * 16 + ccol] = (_Float16)v;
    }
  }
  asm volatile("s_wait_dscnt 0" ::: "memory");

  // ---- layer 2: x2 = relu(x1 @ W2.T + b2), K=64 ----
  v16h xa0 = load_a_lds(xb, H_DIM, 0);
  v16h xa1 = load_a_lds(xb, H_DIM, 32);
  v8f c2[4];
#pragma unroll
  for (int nt = 0; nt < 4; ++nt) {
    v8f acc = {};
    acc = wmma16(xa0, load_b_lds(sW2, H_DIM, nt * 16, 0),  acc);
    acc = wmma16(xa1, load_b_lds(sW2, H_DIM, nt * 16, 32), acc);
    c2[nt] = acc;
  }
  asm volatile("s_wait_dscnt 0" ::: "memory");
#pragma unroll
  for (int nt = 0; nt < 4; ++nt) {
    const float bias = sb2[nt * 16 + ccol];
#pragma unroll
    for (int i = 0; i < 8; ++i) {
      float v = c2[nt][i] + bias;
      v = v > 0.0f ? v : 0.0f;
      xb[(crow + i) * H_DIM + nt * 16 + ccol] = (_Float16)v;
    }
  }
  asm volatile("s_wait_dscnt 0" ::: "memory");

  // ---- gi = x2 @ Wih.T + bih, N=192, K=64 ----
  // A-fragments pulled into registers first, so xb becomes free scratch for
  // restaging C tiles -> wide (b128) coalesced global stores.
  v16h ga0 = load_a_lds(xb, H_DIM, 0);
  v16h ga1 = load_a_lds(xb, H_DIM, 32);
  asm volatile("s_wait_dscnt 0" ::: "memory");
  _Float16* gp = gi + row0 * G_DIM;
  const int srow  = lane >> 1;          // 0..15
  const int shalf = (lane & 1) * 24;    // 0 or 24 (f16 elements)
#pragma unroll
  for (int g = 0; g < 4; ++g) {
#pragma unroll
    for (int q = 0; q < 3; ++q) {
      const int nt = g * 3 + q;
      v8f acc = {};
      acc = wmma16(ga0, load_b_lds(sWih, H_DIM, nt * 16, 0),  acc);
      acc = wmma16(ga1, load_b_lds(sWih, H_DIM, nt * 16, 32), acc);
      const float bias = sbih[nt * 16 + ccol];
#pragma unroll
      for (int i = 0; i < 8; ++i)
        xb[(crow + i) * 48 + q * 16 + ccol] = (_Float16)(acc[i] + bias);
    }
    asm volatile("s_wait_dscnt 0" ::: "memory");
    // cooperative copy of the 16x48 f16 stage to global as 16B vectors
    const _Float16* src = xb + srow * 48 + shalf;
    _Float16* dst = gp + srow * G_DIM + g * 48 + shalf;
#pragma unroll
    for (int c = 0; c < 3; ++c)
      *(v8h*)(dst + c * 8) = *(const v8h*)(src + c * 8);
    asm volatile("s_wait_dscnt 0" ::: "memory");   // reads done before re-stage
  }
}

// =====================================================================
// Kernel 2: persistent GRU scan + fused output head. Per-block 16 batch
// rows (independent recurrences). Time-invariant Whh B-fragments + biases
// hoisted to registers; gi register-double-buffered one step ahead plus
// distance-2 global_prefetch to hide HBM latency on the serial path.
// =====================================================================
__global__ __launch_bounds__(128) void rppo_gru_scan_kernel(
    const float* __restrict__ dones,
    const float* __restrict__ astate, const float* __restrict__ cstate,
    const float* __restrict__ aWhh, const float* __restrict__ abhh,
    const float* __restrict__ cWhh, const float* __restrict__ cbhh,
    const float* __restrict__ aWl, const float* __restrict__ abl,
    const float* __restrict__ cWv, const float* __restrict__ cbv,
    const _Float16* __restrict__ giA, const _Float16* __restrict__ giC,
    float* __restrict__ out_logits, float* __restrict__ out_values,
    float* __restrict__ out_astate, float* __restrict__ out_cstate)
{
  __shared__ _Float16 sWhh[G_DIM * H_DIM];   // 24 KB
  __shared__ float    sbhh[G_DIM];
  __shared__ float    sWl[A_DIM * H_DIM];    // head weights (actor 18x64 / critic 1x64)
  __shared__ float    sbl[A_DIM];
  __shared__ __align__(32) float    sh[16 * H_DIM];    // f32 hidden state
  __shared__ __align__(16) _Float16 shh[16 * H_DIM];   // f16 copy for WMMA A
  __shared__ __align__(32) float    sgh[16 * G_DIM];   // gh result
  __shared__ float    sdone[16];

  const int tid    = threadIdx.x;
  const int branch = blockIdx.y;
  const int b0     = blockIdx.x * 16;

  const float*     Whh   = branch ? cWhh   : aWhh;
  const float*     bhh   = branch ? cbhh   : abhh;
  const _Float16*  gi    = branch ? giC    : giA;
  const float*     state = branch ? cstate : astate;
  float*           outst = branch ? out_cstate : out_astate;
  const int        headN = branch ? 1 : A_DIM;
  const float*     Wh    = branch ? cWv : aWl;
  const float*     bh    = branch ? cbv : abl;

  for (int i = tid; i < G_DIM * H_DIM; i += 128) sWhh[i] = (_Float16)Whh[i];
  for (int i = tid; i < G_DIM; i += 128) sbhh[i] = bhh[i];
  for (int i = tid; i < headN * H_DIM; i += 128) sWl[i] = Wh[i];
  if (tid < headN) sbl[tid] = bh[tid];
  for (int i = tid; i < 16 * H_DIM; i += 128)
    sh[i] = state[(b0 + (i >> 6)) * H_DIM + (i & 63)];
  __syncthreads();

  const int wave = tid >> 5;
  const int lane = tid & 31;
  const int crow = (lane >> 4) << 3;
  const int ccol = lane & 15;

  // ---- hoist time-invariant Whh B-fragments and biases to registers ----
  v16h bf[3][2];
  float bb[3];
#pragma unroll
  for (int q = 0; q < 3; ++q) {
    const int nt = wave * 3 + q;
    bf[q][0] = load_b_lds(sWhh, H_DIM, nt * 16, 0);
    bf[q][1] = load_b_lds(sWhh, H_DIM, nt * 16, 32);
    bb[q]    = sbhh[nt * 16 + ccol];
  }

  // elementwise-phase mapping: 8 contiguous elements per thread
  const int gm = tid >> 3;         // row 0..15
  const int gj = (tid & 7) * 8;    // j0 (multiple of 8)

  // ---- software pipeline: preload dones/gi for t=0 ----
  float dn = (tid < 16) ? dones[b0 + tid] : 0.0f;
  const _Float16* gbase = gi + (long)b0 * G_DIM + gm * G_DIM + gj;
  const long gstep = (long)B_DIM * G_DIM;
  v8h gr_c = *(const v8h*)(gbase);
  v8h gz_c = *(const v8h*)(gbase + H_DIM);
  v8h gn_c = *(const v8h*)(gbase + 2 * H_DIM);

  for (int t = 0; t < T_DIM; ++t) {
    // episode-done reset + quantize h for WMMA
    if (tid < 16) sdone[tid] = dn;
    __syncthreads();
    {
      const float d1 = 1.0f - sdone[gm];
      float* hp = sh + gm * H_DIM + gj;
      _Float16* hq = shh + gm * H_DIM + gj;
#pragma unroll
      for (int i = 0; i < 8; ++i) {
        float hv = hp[i] * d1;
        hp[i] = hv;
        hq[i] = (_Float16)hv;
      }
    }
    // issue next-step loads early (full-iteration overlap window)
    const int tn = (t + 1 < T_DIM) ? t + 1 : t;
    dn = (tid < 16) ? dones[(long)tn * B_DIM + b0 + tid] : 0.0f;
    const _Float16* gnext = gbase + (long)tn * gstep;
    v8h gr_n = *(const v8h*)(gnext);
    v8h gz_n = *(const v8h*)(gnext + H_DIM);
    v8h gn_n = *(const v8h*)(gnext + 2 * H_DIM);
    // distance-2 prefetch into L2 (global_prefetch_b8)
    const int t2 = (t + 2 < T_DIM) ? t + 2 : t;
    const char* pf = (const char*)(gbase + (long)t2 * gstep);
    __builtin_prefetch(pf, 0, 1);
    __builtin_prefetch(pf + 256, 0, 1);
    __syncthreads();

    // gh = h @ Whh.T + bhh  (M=16, N=192, K=64); 3 N-tiles per wave
    v16h ha0 = load_a_lds(shh, H_DIM, 0);
    v16h ha1 = load_a_lds(shh, H_DIM, 32);
#pragma unroll
    for (int q = 0; q < 3; ++q) {
      const int nt = wave * 3 + q;
      v8f acc = {};
      acc = wmma16(ha0, bf[q][0], acc);
      acc = wmma16(ha1, bf[q][1], acc);
#pragma unroll
      for (int i = 0; i < 8; ++i)
        sgh[(crow + i) * G_DIM + nt * 16 + ccol] = acc[i] + bb[q];
    }
    __syncthreads();

    // GRU gating (f32), consuming registers preloaded last iteration
    {
      const float* ghp = sgh + gm * G_DIM + gj;
      float* hp = sh + gm * H_DIM + gj;
#pragma unroll
      for (int i = 0; i < 8; ++i) {
        const float r = 1.0f / (1.0f + __expf(-((float)gr_c[i] + ghp[i])));
        const float z = 1.0f / (1.0f + __expf(-((float)gz_c[i] + ghp[H_DIM + i])));
        const float n = tanhf((float)gn_c[i] + r * ghp[2 * H_DIM + i]);
        hp[i] = (1.0f - z) * n + z * hp[i];
      }
    }
    gr_c = gr_n; gz_c = gz_n; gn_c = gn_n;
    __syncthreads();

    // fused output head
    if (branch == 0) {
      for (int o = tid; o < 16 * A_DIM; o += 128) {
        const int m = o / A_DIM, a = o % A_DIM;
        float acc = sbl[a];
        const float* hr = sh + m * H_DIM;
        const float* wr = sWl + a * H_DIM;
#pragma unroll 8
        for (int j = 0; j < H_DIM; ++j) acc += hr[j] * wr[j];
        out_logits[((long)t * B_DIM + b0 + m) * A_DIM + a] = acc;
      }
    } else {
      if (tid < 16) {
        float acc = sbl[0];
        const float* hr = sh + tid * H_DIM;
#pragma unroll 8
        for (int j = 0; j < H_DIM; ++j) acc += hr[j] * sWl[j];
        out_values[(long)t * B_DIM + b0 + tid] = acc;
      }
    }
    __syncthreads();
  }

  // final recurrent states
  for (int i = tid; i < 16 * H_DIM; i += 128)
    outst[(b0 + (i >> 6)) * H_DIM + (i & 63)] = sh[i];
}

// =====================================================================
extern "C" void kernel_launch(void* const* d_in, const int* in_sizes, int n_in,
                              void* d_out, int out_size, void* d_ws, size_t ws_size,
                              hipStream_t stream) {
  const float* obs    = (const float*)d_in[0];
  const float* dones  = (const float*)d_in[1];
  const float* astate = (const float*)d_in[2];
  const float* cstate = (const float*)d_in[3];
  const float* aW1  = (const float*)d_in[4],  *ab1  = (const float*)d_in[5];
  const float* aW2  = (const float*)d_in[6],  *ab2  = (const float*)d_in[7];
  const float* aWih = (const float*)d_in[8],  *aWhh = (const float*)d_in[9];
  const float* abih = (const float*)d_in[10], *abhh = (const float*)d_in[11];
  const float* aWl  = (const float*)d_in[12], *abl  = (const float*)d_in[13];
  const float* cW1  = (const float*)d_in[14], *cb1  = (const float*)d_in[15];
  const float* cW2  = (const float*)d_in[16], *cb2  = (const float*)d_in[17];
  const float* cWih = (const float*)d_in[18], *cWhh = (const float*)d_in[19];
  const float* cbih = (const float*)d_in[20], *cbhh = (const float*)d_in[21];
  const float* cWv  = (const float*)d_in[22], *cbv  = (const float*)d_in[23];

  float* out         = (float*)d_out;
  float* out_logits  = out;
  float* out_values  = out + (size_t)T_DIM * B_DIM * A_DIM;
  float* out_astate  = out_values + (size_t)T_DIM * B_DIM;
  float* out_cstate  = out_astate + (size_t)B_DIM * H_DIM;

  // workspace: gi (pre-activation input gates) for both branches, f16
  _Float16* giA = (_Float16*)d_ws;
  _Float16* giC = giA + (size_t)T_DIM * B_DIM * G_DIM;

  // Phase 1: massively parallel MLP + gi GEMMs (WMMA)
  dim3 g1((T_DIM * B_DIM) / 64, 2);
  rppo_mlp_gi_kernel<<<g1, 128, 0, stream>>>(
      obs, aW1, ab1, aW2, ab2, aWih, abih,
      cW1, cb1, cW2, cb2, cWih, cbih, giA, giC);

  // Phase 2: persistent per-batch-row GRU scan + fused heads
  dim3 g2(B_DIM / 16, 2);
  rppo_gru_scan_kernel<<<g2, 128, 0, stream>>>(
      dones, astate, cstate, aWhh, abhh, cWhh, cbhh,
      aWl, abl, cWv, cbv, giA, giC,
      out_logits, out_values, out_astate, out_cstate);
}
